// HEGNN_Layer_37606733644279
// MI455X (gfx1250) — matrix operand order repaired and checked by reading, not compiled, not run on hardware
//
#include <hip/hip_runtime.h>

// ---------------------------------------------------------------------------
// HEGNN layer for MI455X (gfx1250): bf16 WMMA tiles, wave32, 16-row tiles.
// Round 2: batched B-fragment loads per K-chunk (overlap loads with WMMA
// bursts), heads computed in-register with xor-shuffle reductions.
// ---------------------------------------------------------------------------

typedef __attribute__((ext_vector_type(16))) __bf16 v16bf;
typedef __attribute__((ext_vector_type(8)))  __bf16 v8bf;
typedef __attribute__((ext_vector_type(8)))  float  v8f;

#define DEVINL __device__ __forceinline__

DEVINL float silu_f(float x) { return x / (1.0f + __expf(-x)); }

// A-fragment (16x32 bf16, M across lanes 0..15, K split per ISA layout):
// lanes 0-15 : elem 0..7 -> K=k0+0..7,  elem 8..15 -> K=k0+16..23
// lanes 16-31: elem 0..7 -> K=k0+8..15, elem 8..15 -> K=k0+24..31
DEVINL v16bf load_a_frag(const __bf16* base, int lda, int k0, int lane) {
  const int row = lane & 15;
  const int hi  = (lane >> 4) & 1;
  const __bf16* p = base + row * lda + k0 + hi * 8;
  union { v16bf v; v8bf h[2]; } u;
  u.h[0] = *(const v8bf*)(p);        // 16B -> ds_load_b128
  u.h[1] = *(const v8bf*)(p + 16);   // 16B -> ds_load_b128
  return u.v;
}

// B fragments are pre-swizzled in workspace: fragment (kt,nt) is 32 lanes x
// 16 bf16 contiguous; lane loads 32 contiguous bytes. All 8 N-tiles of one
// K-chunk are loaded up front so the 8 WMMAs can issue back-to-back.
template <int CHUNKS>
DEVINL void gemm_bf16(const __bf16* A, int lda, const __bf16* Bsw, int lane,
                      v8f acc[8]) {
#pragma unroll
  for (int kt = 0; kt < CHUNKS; ++kt) {
    v16bf a = load_a_frag(A, lda, kt * 32, lane);
    const v16bf* bp = reinterpret_cast<const v16bf*>(Bsw) + (kt * 8) * 32 + lane;
    v16bf b[8];
#pragma unroll
    for (int nt = 0; nt < 8; ++nt) b[nt] = bp[nt * 32];
#pragma unroll
    for (int nt = 0; nt < 8; ++nt)
      acc[nt] = __builtin_amdgcn_wmma_f32_16x16x32_bf16(
          false, a, false, b[nt], (short)0, acc[nt], false, false);
  }
}

DEVINL void zero_acc(v8f acc[8]) {
#pragma unroll
  for (int nt = 0; nt < 8; ++nt)
#pragma unroll
    for (int v = 0; v < 8; ++v) acc[nt][v] = 0.0f;
}

// Select hp[v] for v == ncol (0..7) without dynamic register indexing.
DEVINL float sel8(const float hp[8], int ncol) {
  float s = hp[0];
#pragma unroll
  for (int v = 1; v < 8; ++v)
    if (ncol == v) s = hp[v];
  return s;
}

// ---------------------------------------------------------------------------
// Workspace zeroing
// ---------------------------------------------------------------------------
__global__ void hegnn_zero(float* __restrict__ p, long n) {
  long i = (long)blockIdx.x * blockDim.x + threadIdx.x;
  long stride = (long)gridDim.x * blockDim.x;
  for (; i < n; i += stride) p[i] = 0.0f;
}

// ---------------------------------------------------------------------------
// Weight conversion fp32 -> bf16, pre-swizzled into WMMA B-fragment order.
// dst[((kt*8 + nt)*32 + lane)*16 + i] = W[k][n],
//   k = kt*32 + (lane>=16 ? 8 : 0) + (i<8 ? i : 16 + i - 8),  n = nt*16 + (lane&15)
// ---------------------------------------------------------------------------
__global__ void hegnn_swizzle_w(const float* __restrict__ W,
                                __bf16* __restrict__ dst, int Kreal,
                                int chunks) {
  int idx = blockIdx.x * blockDim.x + threadIdx.x;
  int total = chunks * 8 * 32 * 16;
  if (idx >= total) return;
  int i = idx & 15;
  int lane = (idx >> 4) & 31;
  int nt = (idx >> 9) & 7;
  int kt = idx >> 12;
  int k = kt * 32 + ((lane >> 4) & 1) * 8 + (i < 8 ? i : 16 + (i - 8));
  int n = nt * 16 + (lane & 15);
  float v = (k < Kreal) ? W[k * 128 + n] : 0.0f;
  dst[idx] = (__bf16)v;
}

// ---------------------------------------------------------------------------
// Edge kernel: 4 waves/block, one 16-edge tile per wave.
// ---------------------------------------------------------------------------
__global__ __launch_bounds__(128, 1) void hegnn_edge_kernel(
    const float* __restrict__ feat, const float* __restrict__ pos,
    const float* __restrict__ shv, const float* __restrict__ eattr,
    const int* __restrict__ erow, const int* __restrict__ ecol,
    const float* __restrict__ bm1, const float* __restrict__ bm2,
    const float* __restrict__ bp1, const float* __restrict__ Wp2,
    const float* __restrict__ bp2, const float* __restrict__ bs1,
    const float* __restrict__ Ws2, const float* __restrict__ bs2,
    const __bf16* __restrict__ Wm1s, const __bf16* __restrict__ Wm2s,
    const __bf16* __restrict__ Wp1s, const __bf16* __restrict__ Ws1s,
    float* __restrict__ cnt, float* __restrict__ msg_agg,
    float* __restrict__ pos_agg, float* __restrict__ sh_agg, int E) {
  constexpr int K1P = 288;                         // 277 padded to 9 K-chunks
  constexpr int WB = K1P * 16 * 2 + 16 * 128 * 2;  // 13312 B per wave
  __shared__ __align__(16) char smem[4 * WB];

  const int wave = threadIdx.x >> 5;
  const int lane = threadIdx.x & 31;
  char* wb = smem + wave * WB;
  __bf16* sX = (__bf16*)wb;                   // [16][288] bf16 input tile
  __bf16* sM = (__bf16*)(wb + K1P * 16 * 2);  // [16][128] bf16 hidden/msg

  const int tile = blockIdx.x * 4 + wave;
  const int e0 = tile * 16;
  if (e0 >= E) return;

  __builtin_prefetch(Wm1s + (lane << 6), 0, 0);  // global_prefetch weights

  // ---- build edge-MLP input tile X[16][288] in LDS (bf16) -----------------
  for (int idx = lane; idx < 16 * K1P; idx += 32) {
    int el = idx / K1P, c = idx - el * K1P;
    int e = e0 + el;
    if (e >= E) e = E - 1;
    int r = erow[e], cc = ecol[e];
    float val = 0.0f;
    if (c < 128) {
      val = feat[r * 128 + c];
    } else if (c < 256) {
      val = feat[cc * 128 + (c - 128)];
    } else if (c < 272) {
      val = eattr[e * 16 + (c - 256)];
    } else if (c == 272) {  // squared distance
      float s = 0.0f;
#pragma unroll
      for (int d = 0; d < 3; ++d) {
        float dd = pos[r * 3 + d] - pos[cc * 3 + d];
        s += dd * dd;
      }
      val = s;
    } else if (c < 277) {  // per-degree SH inner products
      int l = c - 273, st = l * l, en = (l + 1) * (l + 1);
      float s = 0.0f;
      for (int j = st; j < en; ++j) s += shv[r * 16 + j] * shv[cc * 16 + j];
      val = s;
    }
    sX[idx] = (__bf16)val;
  }

  const int mrow = (lane >> 4) * 8;  // C/D: VGPR v holds M = mrow + v
  const int ncol = lane & 15;        //      N = nt*16 + ncol

  // ---- msg MLP layer 1: [16x288] @ [288x128] ------------------------------
  v8f acc[8];
  zero_acc(acc);
  gemm_bf16<9>(sX, K1P, Wm1s, lane, acc);
#pragma unroll
  for (int nt = 0; nt < 8; ++nt) {
    int n = nt * 16 + ncol;
    float b = bm1[n];
#pragma unroll
    for (int v = 0; v < 8; ++v)
      sM[(mrow + v) * 128 + n] = (__bf16)silu_f(acc[nt][v] + b);
  }

  // ---- msg MLP layer 2 + scatter msg, count -------------------------------
  int rs[8];
#pragma unroll
  for (int v = 0; v < 8; ++v) {
    int e = e0 + mrow + v;
    rs[v] = erow[e < E ? e : (E - 1)];
  }
  zero_acc(acc);
  gemm_bf16<4>(sM, 128, Wm2s, lane, acc);
#pragma unroll
  for (int nt = 0; nt < 8; ++nt) {
    int n = nt * 16 + ncol;
    float b = bm2[n];
#pragma unroll
    for (int v = 0; v < 8; ++v) {
      float m = silu_f(acc[nt][v] + b);
      sM[(mrow + v) * 128 + n] = (__bf16)m;  // msg (bf16) for pos/sh MLPs
      if (e0 + mrow + v < E) atomicAdd(&msg_agg[rs[v] * 128 + n], m);
    }
  }
  if (lane < 16 && e0 + lane < E) atomicAdd(&cnt[erow[e0 + lane]], 1.0f);

  // ---- pos gate MLP: hidden via WMMA, head via in-register reduction ------
  zero_acc(acc);
  gemm_bf16<4>(sM, 128, Wp1s, lane, acc);
  {
    float hp[8];
#pragma unroll
    for (int v = 0; v < 8; ++v) hp[v] = 0.0f;
#pragma unroll
    for (int nt = 0; nt < 8; ++nt) {
      int n = nt * 16 + ncol;
      float b = bp1[n];
      float w = Wp2[n];
#pragma unroll
      for (int v = 0; v < 8; ++v) hp[v] += silu_f(acc[nt][v] + b) * w;
    }
    // xor reduce within each 16-lane half (each half holds all 128 channels)
#pragma unroll
    for (int off = 1; off < 16; off <<= 1)
#pragma unroll
      for (int v = 0; v < 8; ++v) hp[v] += __shfl_xor(hp[v], off, 32);
    if (ncol < 8) {
      int e = e0 + mrow + ncol;
      if (e < E) {
        int r = erow[e], cc = ecol[e];
        float s = sel8(hp, ncol) + bp2[0];
#pragma unroll
        for (int d = 0; d < 3; ++d)
          atomicAdd(&pos_agg[r * 3 + d], (pos[r * 3 + d] - pos[cc * 3 + d]) * s);
      }
    }
  }

  // ---- sh gate MLP: hidden via WMMA, 4-way head, expand by degree ---------
  zero_acc(acc);
  gemm_bf16<4>(sM, 128, Ws1s, lane, acc);
  {
    float pg[4][8];
#pragma unroll
    for (int l = 0; l < 4; ++l)
#pragma unroll
      for (int v = 0; v < 8; ++v) pg[l][v] = 0.0f;
#pragma unroll
    for (int nt = 0; nt < 8; ++nt) {
      int n = nt * 16 + ncol;
      float b = bs1[n];
      float w0 = Ws2[n * 4 + 0], w1 = Ws2[n * 4 + 1];
      float w2 = Ws2[n * 4 + 2], w3 = Ws2[n * 4 + 3];
#pragma unroll
      for (int v = 0; v < 8; ++v) {
        float h = silu_f(acc[nt][v] + b);
        pg[0][v] += h * w0;
        pg[1][v] += h * w1;
        pg[2][v] += h * w2;
        pg[3][v] += h * w3;
      }
    }
#pragma unroll
    for (int off = 1; off < 16; off <<= 1)
#pragma unroll
      for (int l = 0; l < 4; ++l)
#pragma unroll
        for (int v = 0; v < 8; ++v) pg[l][v] += __shfl_xor(pg[l][v], off, 32);
    if (ncol < 8) {
      int e = e0 + mrow + ncol;
      if (e < E) {
        int r = erow[e], cc = ecol[e];
        float g[4];
#pragma unroll
        for (int l = 0; l < 4; ++l) g[l] = sel8(pg[l], ncol) + bs2[l];
        const int DEG[16] = {0, 1, 1, 1, 2, 2, 2, 2, 2, 3, 3, 3, 3, 3, 3, 3};
#pragma unroll
        for (int m = 0; m < 16; ++m)
          atomicAdd(&sh_agg[r * 16 + m],
                    (shv[r * 16 + m] - shv[cc * 16 + m]) * g[DEG[m]]);
      }
    }
  }
}

// ---------------------------------------------------------------------------
// Node kernel: 4 waves/block, one 16-node tile per wave.
// ---------------------------------------------------------------------------
__global__ __launch_bounds__(128, 1) void hegnn_node_kernel(
    const float* __restrict__ feat, const float* __restrict__ pos,
    const float* __restrict__ shv, const float* __restrict__ vel,
    const float* __restrict__ bv1, const float* __restrict__ Wv2,
    const float* __restrict__ bv2, const float* __restrict__ bn1,
    const float* __restrict__ bn2, const __bf16* __restrict__ Wv1s,
    const __bf16* __restrict__ Wn1s, const __bf16* __restrict__ Wn2s,
    const float* __restrict__ cnt, const float* __restrict__ msg_agg,
    const float* __restrict__ pos_agg, const float* __restrict__ sh_agg,
    float* __restrict__ out_feat, float* __restrict__ out_pos,
    float* __restrict__ out_sh, int N) {
  constexpr int WB = 16 * 256 * 2 + 16 * 128 * 2;  // 12288 B per wave
  __shared__ __align__(16) char smem[4 * WB];

  const int wave = threadIdx.x >> 5;
  const int lane = threadIdx.x & 31;
  char* wb = smem + wave * WB;
  __bf16* sA = (__bf16*)wb;           // [16][256]: [feat | msg_agg/denom]
  __bf16* sB = (__bf16*)(wb + 8192);  // [16][128] bf16 hidden

  const int tile = blockIdx.x * 4 + wave;
  const int n0 = tile * 16;
  if (n0 >= N) return;

  for (int idx = lane; idx < 16 * 256; idx += 32) {
    int nl = idx >> 8, c = idx & 255;
    int node = n0 + nl;
    if (node >= N) node = N - 1;
    float val;
    if (c < 128) {
      val = feat[node * 128 + c];
    } else {
      float d = fmaxf(cnt[node], 1.0f);
      val = msg_agg[node * 128 + (c - 128)] / d;
    }
    sA[idx] = (__bf16)val;
  }

  const int mrow = (lane >> 4) * 8;
  const int ncol = lane & 15;

  // ---- vel MLP: 128->128 silu (WMMA), head 128->1, pos update -------------
  v8f acc[8];
  zero_acc(acc);
  gemm_bf16<4>(sA, 256, Wv1s, lane, acc);  // first 128 cols of sA = feat
  {
    float hp[8];
#pragma unroll
    for (int v = 0; v < 8; ++v) hp[v] = 0.0f;
#pragma unroll
    for (int nt = 0; nt < 8; ++nt) {
      int n = nt * 16 + ncol;
      float b = bv1[n];
      float w = Wv2[n];
#pragma unroll
      for (int v = 0; v < 8; ++v) hp[v] += silu_f(acc[nt][v] + b) * w;
    }
#pragma unroll
    for (int off = 1; off < 16; off <<= 1)
#pragma unroll
      for (int v = 0; v < 8; ++v) hp[v] += __shfl_xor(hp[v], off, 32);
    if (ncol < 8) {
      int node = n0 + mrow + ncol;
      if (node < N) {
        float s = sel8(hp, ncol) + bv2[0];
        float d = fmaxf(cnt[node], 1.0f);
#pragma unroll
        for (int dd = 0; dd < 3; ++dd)
          out_pos[node * 3 + dd] = pos[node * 3 + dd] +
                                   pos_agg[node * 3 + dd] / d +
                                   s * vel[node * 3 + dd];
      }
    }
  }

  // ---- new_sh -------------------------------------------------------------
  for (int idx = lane; idx < 256; idx += 32) {
    int nl = idx >> 4, m = idx & 15;
    int node = n0 + nl;
    if (node < N) {
      float d = fmaxf(cnt[node], 1.0f);
      out_sh[node * 16 + m] = shv[node * 16 + m] + sh_agg[node * 16 + m] / d;
    }
  }

  // ---- node MLP: 256->128 silu, 128->128 ----------------------------------
  zero_acc(acc);
  gemm_bf16<8>(sA, 256, Wn1s, lane, acc);
#pragma unroll
  for (int nt = 0; nt < 8; ++nt) {
    int n = nt * 16 + ncol;
    float b = bn1[n];
#pragma unroll
    for (int v = 0; v < 8; ++v)
      sB[(mrow + v) * 128 + n] = (__bf16)silu_f(acc[nt][v] + b);
  }
  zero_acc(acc);
  gemm_bf16<4>(sB, 128, Wn2s, lane, acc);
#pragma unroll
  for (int nt = 0; nt < 8; ++nt) {
    int n = nt * 16 + ncol;
    float b = bn2[n];
#pragma unroll
    for (int v = 0; v < 8; ++v) {
      int node = n0 + mrow + v;
      if (node < N) out_feat[node * 128 + n] = acc[nt][v] + b;
    }
  }
}

// ---------------------------------------------------------------------------
// Host launcher
// ---------------------------------------------------------------------------
extern "C" void kernel_launch(void* const* d_in, const int* in_sizes, int n_in,
                              void* d_out, int out_size, void* d_ws,
                              size_t ws_size, hipStream_t stream) {
  (void)in_sizes; (void)n_in; (void)out_size; (void)ws_size;
  const int N = 50000, E = 800000;

  const float* feat = (const float*)d_in[0];
  const float* pos = (const float*)d_in[1];
  const float* shv = (const float*)d_in[2];
  const float* vel = (const float*)d_in[3];
  const float* eattr = (const float*)d_in[4];
  const int* eidx = (const int*)d_in[5];
  const int* erow = eidx;
  const int* ecol = eidx + E;
  const float* Wm1 = (const float*)d_in[6];
  const float* bm1 = (const float*)d_in[7];
  const float* Wm2 = (const float*)d_in[8];
  const float* bm2 = (const float*)d_in[9];
  const float* Wp1 = (const float*)d_in[10];
  const float* bp1 = (const float*)d_in[11];
  const float* Wp2 = (const float*)d_in[12];
  const float* bp2 = (const float*)d_in[13];
  const float* Ws1 = (const float*)d_in[14];
  const float* bs1 = (const float*)d_in[15];
  const float* Ws2 = (const float*)d_in[16];
  const float* bs2 = (const float*)d_in[17];
  const float* Wv1 = (const float*)d_in[18];
  const float* bv1 = (const float*)d_in[19];
  const float* Wv2 = (const float*)d_in[20];
  const float* bv2 = (const float*)d_in[21];
  const float* Wn1 = (const float*)d_in[22];
  const float* bn1 = (const float*)d_in[23];
  const float* Wn2 = (const float*)d_in[24];
  const float* bn2 = (const float*)d_in[25];

  // workspace layout: f32 accumulators, then bf16 swizzled weights
  float* cnt = (float*)d_ws;
  float* msg_agg = cnt + N;                // N*128
  float* pos_agg = cnt + (size_t)N * 129;  // N*3
  float* sh_agg = cnt + (size_t)N * 132;   // N*16
  size_t accBytes = (size_t)N * 148 * 4;
  char* wp = (char*)d_ws + ((accBytes + 255) & ~(size_t)255);
  __bf16* Wm1s = (__bf16*)wp; wp += (size_t)9 * 4096 * 2;
  __bf16* Wm2s = (__bf16*)wp; wp += (size_t)4 * 4096 * 2;
  __bf16* Wp1s = (__bf16*)wp; wp += (size_t)4 * 4096 * 2;
  __bf16* Ws1s = (__bf16*)wp; wp += (size_t)4 * 4096 * 2;
  __bf16* Wv1s = (__bf16*)wp; wp += (size_t)4 * 4096 * 2;
  __bf16* Wn1s = (__bf16*)wp; wp += (size_t)8 * 4096 * 2;
  __bf16* Wn2s = (__bf16*)wp; wp += (size_t)4 * 4096 * 2;

  float* out_feat = (float*)d_out;
  float* out_pos = out_feat + (size_t)N * 128;
  float* out_sh = out_pos + (size_t)N * 3;

  // 1) zero scatter accumulators (must happen every call)
  long zn = (long)N * 148;
  hegnn_zero<<<4096, 256, 0, stream>>>(cnt, zn);

  // 2) convert + swizzle weights into WMMA B-fragment order (bf16)
  hegnn_swizzle_w<<<(9 * 4096 + 255) / 256, 256, 0, stream>>>(Wm1, Wm1s, 277, 9);
  hegnn_swizzle_w<<<(4 * 4096 + 255) / 256, 256, 0, stream>>>(Wm2, Wm2s, 128, 4);
  hegnn_swizzle_w<<<(4 * 4096 + 255) / 256, 256, 0, stream>>>(Wp1, Wp1s, 128, 4);
  hegnn_swizzle_w<<<(4 * 4096 + 255) / 256, 256, 0, stream>>>(Ws1, Ws1s, 128, 4);
  hegnn_swizzle_w<<<(4 * 4096 + 255) / 256, 256, 0, stream>>>(Wv1, Wv1s, 128, 4);
  hegnn_swizzle_w<<<(8 * 4096 + 255) / 256, 256, 0, stream>>>(Wn1, Wn1s, 256, 8);
  hegnn_swizzle_w<<<(4 * 4096 + 255) / 256, 256, 0, stream>>>(Wn2, Wn2s, 128, 4);

  // 3) edge pipeline: msg/pos/sh MLPs (WMMA) + scatter atomics
  int etiles = (E + 15) / 16;
  hegnn_edge_kernel<<<(etiles + 3) / 4, 128, 0, stream>>>(
      feat, pos, shv, eattr, erow, ecol, bm1, bm2, bp1, Wp2, bp2, bs1, Ws2,
      bs2, Wm1s, Wm2s, Wp1s, Ws1s, cnt, msg_agg, pos_agg, sh_agg, E);

  // 4) node update: scatter-mean finalize + vel/node MLPs (WMMA)
  int ntiles = (N + 15) / 16;
  hegnn_node_kernel<<<(ntiles + 3) / 4, 128, 0, stream>>>(
      feat, pos, shv, vel, bv1, Wv2, bv2, bn1, bn2, Wv1s, Wn1s, Wn2s, cnt,
      msg_agg, pos_agg, sh_agg, out_feat, out_pos, out_sh, N);
}
